// MultiHeadAttention_80839874445624
// MI455X (gfx1250) — matrix operand order repaired and checked by reading, not compiled, not run on hardware
//
#include <hip/hip_runtime.h>
#include <hip/hip_bf16.h>
#include <hip/hip_fp16.h>

#define B_  2
#define S_  2048
#define D_  1024
#define H_  16
#define DH_ 64

typedef __attribute__((ext_vector_type(16))) _Float16 v16h;
typedef __attribute__((ext_vector_type(8)))  _Float16 v8h;
typedef __attribute__((ext_vector_type(8)))  float    v8f;

static __device__ __forceinline__ v16h cat8(v8h lo, v8h hi) {
  return __builtin_shufflevector(lo, hi, 0,1,2,3,4,5,6,7,8,9,10,11,12,13,14,15);
}
static __device__ __forceinline__ v8f wmma_f16(v16h a, v16h b, v8f c) {
  // D = A(16x32 f16) * B(32x16 f16) + C(16x16 f32)
  return __builtin_amdgcn_wmma_f32_16x16x32_f16(false, a, false, b, (short)0, c,
                                                false, false);
}
// A-fragment: two contiguous 16B chunks at p and p+8 halves
static __device__ __forceinline__ v16h loadA(const _Float16* p) {
  return cat8(*(const v8h*)p, *(const v8h*)(p + 8));
}

// ---------------------------------------------------------------------------
// fp32 -> fp16 elementwise convert (grid-stride)
// ---------------------------------------------------------------------------
__global__ void cvt_f32_to_f16(const float* __restrict__ src,
                               _Float16* __restrict__ dst, int n) {
  int i = blockIdx.x * blockDim.x + threadIdx.x;
  int stride = gridDim.x * blockDim.x;
  for (; i < n; i += stride) dst[i] = (_Float16)src[i];
}

// ---------------------------------------------------------------------------
// C[M,N] = A[M,K] (f16 row-major) * W[N,K]^T (f16 row-major) + bias[N]
//  mode 0: fp32 out, row-major [M][N]            (final projection)
//  mode 1: f16  out, (B,H,S,DH) layout           (Q, K)
//  mode 2: f16  out, (B,H,DH,S) layout           (V transposed)
// One wave per 16(M) x 64(N) tile. Explicitly double-buffered k-loop:
// fragments for step k+32 are loaded into a second register set before the
// step-k WMMAs consume the first set, so global-load latency hides behind
// 4 WMMAs instead of stalling each one (s_wait_loadcnt != 0x0).
// Fragment loads use contiguous k order, identical for A and B:
//  A-frag: lane holds row (lane&15); halves 0..15 -> k = aoff+h
//  B-frag: lane holds col (lane&15); halves 0..15 -> k = boff+h
//  C-frag: lane holds col (lane&15); VGPR r -> row r + 8*(lane>=16)
// ---------------------------------------------------------------------------
__global__ void gemm_nt_wmma(const _Float16* __restrict__ A,
                             const _Float16* __restrict__ W,
                             const float* __restrict__ bias,
                             void* __restrict__ out,
                             int M, int N, int K, int mode) {
  const int lane  = threadIdx.x;
  const int mtile = blockIdx.y * blockDim.y + threadIdx.y;
  const int n0    = blockIdx.x * 64;         // four 16-wide N tiles
  const int ln    = lane & 15;
  const int hl    = lane >> 4;               // half-wave select
  const size_t wstep = (size_t)16 * (size_t)K;   // next 16-col tile of W
  const _Float16* Ap = A + (size_t)(mtile * 16 + ln) * (size_t)K + hl * 16;
  const _Float16* Wp = W + (size_t)(n0 + ln) * (size_t)K + hl * 16;

  v8f acc[4] = {{}, {}, {}, {}};
  v16h af0, af1, bf0[4], bf1[4];

  // prologue: stage k = 0
  af0 = loadA(Ap);
#pragma unroll
  for (int t = 0; t < 4; ++t) bf0[t] = *(const v16h*)(Wp + t * wstep);

  for (int k = 0; k < K; k += 64) {
    // stage k+32 while consuming k
    af1 = loadA(Ap + k + 32);
#pragma unroll
    for (int t = 0; t < 4; ++t)
      bf1[t] = *(const v16h*)(Wp + t * wstep + k + 32);
    __builtin_prefetch(Ap + k + 512, 0, 0);        // global_prefetch_b8
#pragma unroll
    for (int t = 0; t < 4; ++t) acc[t] = wmma_f16(af0, bf0[t], acc[t]);

    // stage k+64 while consuming k+32
    if (k + 64 < K) {
      af0 = loadA(Ap + k + 64);
#pragma unroll
      for (int t = 0; t < 4; ++t)
        bf0[t] = *(const v16h*)(Wp + t * wstep + k + 64);
    }
#pragma unroll
    for (int t = 0; t < 4; ++t) acc[t] = wmma_f16(af1, bf1[t], acc[t]);
  }

#pragma unroll
  for (int t = 0; t < 4; ++t) {
    const int ncol = n0 + t * 16 + ln;
    const float bv = bias ? bias[ncol] : 0.0f;
    if (mode == 0) {
      float* o = (float*)out;
#pragma unroll
      for (int r = 0; r < 8; ++r) {
        int m = mtile * 16 + r + hl * 8;
        o[(size_t)m * (size_t)N + ncol] = acc[t][r] + bv;
      }
    } else {
      _Float16* o = (_Float16*)out;
      const int h = ncol / DH_, dh = ncol % DH_;
#pragma unroll
      for (int r = 0; r < 8; ++r) {
        int m = mtile * 16 + r + hl * 8;
        int b = m / S_, s = m % S_;
        float v = acc[t][r] + bv;
        if (mode == 1)
          o[(((size_t)(b * H_ + h)) * S_ + s) * DH_ + dh] = (_Float16)v;
        else
          o[(((size_t)(b * H_ + h)) * DH_ + dh) * S_ + s] = (_Float16)v;
      }
    }
  }
}

// ---------------------------------------------------------------------------
// Flash attention: one wave owns (b,h, 16-query tile). Keys stream in steps
// of 32 (two 16x16 score C-frags). Online softmax in registers; P panel
// staged through 1KB LDS to re-layout C-frag -> A-frag for the PV WMMA.
// Q,K: (B,H,S,DH) f16.  VT: (B,H,DH,S) f16.  AO out: (B,S,H*DH) f16.
// ---------------------------------------------------------------------------
__global__ void flash_attn_wmma(const _Float16* __restrict__ Q,
                                const _Float16* __restrict__ Kmat,
                                const _Float16* __restrict__ VT,
                                _Float16* __restrict__ AO) {
  __shared__ _Float16 plds[4][16 * 32];
  const int lane   = threadIdx.x;
  const int wv     = threadIdx.y;
  const int waveId = blockIdx.x * 4 + wv;
  const int qt = waveId & (S_ / 16 - 1);      // 128 query tiles per (b,h)
  const int bh = waveId >> 7;
  const int ln = lane & 15;
  const int hl = lane >> 4;

  const _Float16* Qb = Q    + (size_t)bh * S_ * DH_;
  const _Float16* Kb = Kmat + (size_t)bh * S_ * DH_;
  const _Float16* Vb = VT   + (size_t)bh * DH_ * S_;
  _Float16* lp = plds[wv];

  // Q A-fragments for the two 32-wide k-steps of DH=64
  const size_t qrow = (size_t)(qt * 16 + ln) * DH_;
  v16h qa[2];
#pragma unroll
  for (int d = 0; d < 2; ++d)
    qa[d] = loadA(Qb + qrow + d * 32 + hl * 16);

  float rmax[8], rsum[8];
  v8f o0 = {}, o1 = {}, o2 = {}, o3 = {};
#pragma unroll
  for (int r = 0; r < 8; ++r) { rmax[r] = -3.0e38f; rsum[r] = 0.0f; }

  const float scale = 0.125f;                 // 1/sqrt(64)
  const int keyEnd = qt * 16 + 16;            // causal: keys t <= last query

  for (int j = 0; j < keyEnd; j += 32) {
    // ---- scores: two 16x16 tiles (cols j..j+15, j+16..j+31) ----
    v8f s0 = {}, s1 = {};
    const _Float16* k0 = Kb + (size_t)(j + ln) * DH_;
    const _Float16* k1 = Kb + (size_t)(j + 16 + ln) * DH_;
#pragma unroll
    for (int d = 0; d < 2; ++d) {
      s0 = wmma_f16(qa[d], *(const v16h*)(k0 + d * 32 + hl * 16), s0);
      s1 = wmma_f16(qa[d], *(const v16h*)(k1 + d * 32 + hl * 16), s1);
    }

    // ---- scale + causal mask + online softmax (C-frag layout) ----
    const int t0 = j + ln, t1 = j + 16 + ln;
#pragma unroll
    for (int r = 0; r < 8; ++r) {
      const int q = qt * 16 + r + hl * 8;
      float a = s0[r] * scale; if (t0 > q) a = -1.0e30f;
      float b = s1[r] * scale; if (t1 > q) b = -1.0e30f;
      // row max across the 16 lanes holding this row (xor bits 0..3 only)
      float v = fmaxf(a, b);
      v = fmaxf(v, __shfl_xor(v, 1, 32));
      v = fmaxf(v, __shfl_xor(v, 2, 32));
      v = fmaxf(v, __shfl_xor(v, 4, 32));
      v = fmaxf(v, __shfl_xor(v, 8, 32));
      float mnew  = fmaxf(rmax[r], v);
      float alpha = __expf(rmax[r] - mnew);
      float p0 = __expf(a - mnew);
      float p1 = __expf(b - mnew);
      float ps = p0 + p1;
      ps += __shfl_xor(ps, 1, 32);
      ps += __shfl_xor(ps, 2, 32);
      ps += __shfl_xor(ps, 4, 32);
      ps += __shfl_xor(ps, 8, 32);
      rsum[r] = rsum[r] * alpha + ps;
      rmax[r] = mnew;
      o0[r] *= alpha; o1[r] *= alpha; o2[r] *= alpha; o3[r] *= alpha;
      // stage P (f16) into LDS in plain [16][32] layout
      const int m = r + hl * 8;
      lp[m * 32 + ln]      = (_Float16)p0;
      lp[m * 32 + 16 + ln] = (_Float16)p1;
    }
    __builtin_amdgcn_wave_barrier();
    asm volatile("s_wait_dscnt 0" ::: "memory");  // DS in-order per wave; belt+suspenders

    // ---- re-read P as A-fragment (16x32, k = key-offset) ----
    v16h pa = loadA(lp + ln * 32 + hl * 16);

    // ---- PV: B-frag from V^T rows are contiguous along S ----
    const size_t vcol = (size_t)j + hl * 16;
    o0 = wmma_f16(pa, *(const v16h*)(Vb + (size_t)(0 * 16 + ln) * S_ + vcol), o0);
    o1 = wmma_f16(pa, *(const v16h*)(Vb + (size_t)(1 * 16 + ln) * S_ + vcol), o1);
    o2 = wmma_f16(pa, *(const v16h*)(Vb + (size_t)(2 * 16 + ln) * S_ + vcol), o2);
    o3 = wmma_f16(pa, *(const v16h*)(Vb + (size_t)(3 * 16 + ln) * S_ + vcol), o3);
  }

  // ---- normalize and write AO in (B,S,H*DH) f16 for the output GEMM ----
  const int b = bh >> 4;          // bh / H
  const int hbase = (bh & (H_ - 1)) * DH_;
#pragma unroll
  for (int r = 0; r < 8; ++r) {
    const int s = qt * 16 + r + hl * 8;
    const float inv = 1.0f / rsum[r];
    const size_t base = ((size_t)(b * S_ + s)) * D_ + hbase;
    AO[base + 0 * 16 + ln] = (_Float16)(o0[r] * inv);
    AO[base + 1 * 16 + ln] = (_Float16)(o1[r] * inv);
    AO[base + 2 * 16 + ln] = (_Float16)(o2[r] * inv);
    AO[base + 3 * 16 + ln] = (_Float16)(o3[r] * inv);
  }
}

// ---------------------------------------------------------------------------
extern "C" void kernel_launch(void* const* d_in, const int* in_sizes, int n_in,
                              void* d_out, int out_size, void* d_ws, size_t ws_size,
                              hipStream_t stream) {
  (void)in_sizes; (void)n_in; (void)out_size; (void)ws_size;
  const float* x  = (const float*)d_in[0];
  // d_in[1] = causal mask (known analytically, unused)
  const float* Wq = (const float*)d_in[2];
  const float* bq = (const float*)d_in[3];
  const float* Wk = (const float*)d_in[4];
  const float* bk = (const float*)d_in[5];
  const float* Wv = (const float*)d_in[6];
  const float* bv = (const float*)d_in[7];
  const float* Wo = (const float*)d_in[8];
  const float* bo = (const float*)d_in[9];

  char* ws = (char*)d_ws;
  const size_t MB = 1024 * 1024;
  _Float16* xh  = (_Float16*)(ws + 0);         // 8 MiB  (B*S*D f16)
  _Float16* wqh = (_Float16*)(ws + 8  * MB);   // 2 MiB each
  _Float16* wkh = (_Float16*)(ws + 10 * MB);
  _Float16* wvh = (_Float16*)(ws + 12 * MB);
  _Float16* woh = (_Float16*)(ws + 14 * MB);
  _Float16* Qb  = (_Float16*)(ws + 16 * MB);   // (B,H,S,DH)
  _Float16* Kb  = (_Float16*)(ws + 24 * MB);   // (B,H,S,DH)
  _Float16* VTb = (_Float16*)(ws + 32 * MB);   // (B,H,DH,S)
  _Float16* AOb = (_Float16*)(ws + 40 * MB);   // (B,S,D)     total 48 MiB

  const int M = B_ * S_;                       // 4096
  const int nx = B_ * S_ * D_;                 // 4,194,304
  const int nw = D_ * D_;                      // 1,048,576

  cvt_f32_to_f16<<<4096, 256, 0, stream>>>(x,  xh,  nx);
  cvt_f32_to_f16<<<2048, 256, 0, stream>>>(Wq, wqh, nw);
  cvt_f32_to_f16<<<2048, 256, 0, stream>>>(Wk, wkh, nw);
  cvt_f32_to_f16<<<2048, 256, 0, stream>>>(Wv, wvh, nw);
  cvt_f32_to_f16<<<2048, 256, 0, stream>>>(Wo, woh, nw);

  dim3 gblk(32, 4);                             // 4 waves per block
  dim3 ggrd(D_ / 64, M / 64);                   // 16x64 tile per wave
  gemm_nt_wmma<<<ggrd, gblk, 0, stream>>>(xh, wqh, bq, Qb,  M, D_, D_, 1);
  gemm_nt_wmma<<<ggrd, gblk, 0, stream>>>(xh, wkh, bk, Kb,  M, D_, D_, 1);
  gemm_nt_wmma<<<ggrd, gblk, 0, stream>>>(xh, wvh, bv, VTb, M, D_, D_, 2);

  // B*H*(S/16) = 4096 waves, 4 per block
  flash_attn_wmma<<<(B_ * H_ * (S_ / 16)) / 4, gblk, 0, stream>>>(Qb, Kb, VTb, AOb);

  gemm_nt_wmma<<<ggrd, gblk, 0, stream>>>(AOb, woh, bo, d_out, M, D_, D_, 0);
}
